// ChebConv_47339129536658
// MI455X (gfx1250) — matrix operand order repaired and checked by reading, not compiled, not run on hardware
//
#include <hip/hip_runtime.h>

typedef _Float16 half_t;
typedef __attribute__((ext_vector_type(16))) _Float16 v16h;
typedef __attribute__((ext_vector_type(8)))  _Float16 v8h;
typedef __attribute__((ext_vector_type(8)))  float    v8f;
typedef __attribute__((ext_vector_type(4)))  float    v4f;
typedef __attribute__((ext_vector_type(2)))  float    v2f;

#define BB    4
#define NN    4096
#define CC    128
#define KCH   3
#define FDIM  (KCH * CC)                // 384
#define LRC   ((float)(2.0 / (2.0 + 1e-06)))
#define DCHNK 256                       // rows per degree partial block
#define NCHNK (NN / DCHNK)              // 16 partials per column

// ---------------------------------------------------------------------------
// Stage 1: partial column sums over a 256-row chunk.
// grid = (NN/256, NCHNK, BB), block = 256.  1024 blocks -> saturates HBM.
// ---------------------------------------------------------------------------
__global__ __launch_bounds__(256)
void degree_partial_kernel(const float* __restrict__ adj,
                           float* __restrict__ partial) {
    const int b     = blockIdx.z;
    const int chunk = blockIdx.y;
    const int m     = blockIdx.x * 256 + threadIdx.x;
    const float* p  = adj + (size_t)b * NN * NN + (size_t)chunk * DCHNK * NN + m;
    float s = 0.0f;
    #pragma unroll 8
    for (int n = 0; n < DCHNK; ++n) s += p[(size_t)n * NN];
    partial[((size_t)b * NCHNK + chunk) * NN + m] = s;
}

// Stage 2: reduce the NCHNK partials.  grid = (NN/256, BB).
__global__ __launch_bounds__(256)
void degree_reduce_kernel(const float* __restrict__ partial,
                          float* __restrict__ degree) {
    const int b = blockIdx.y;
    const int m = blockIdx.x * 256 + threadIdx.x;
    float s = 0.0f;
    #pragma unroll
    for (int c = 0; c < NCHNK; ++c)
        s += partial[((size_t)b * NCHNK + c) * NN + m];
    degree[b * NN + m] = s;
}

// ---------------------------------------------------------------------------
// Tiled transpose: src f32 [N][C] (per batch) -> dstT f16 [C][N].
// Optionally also writes an f32 copy (used to place Z0 = x into the Z buffer).
// ---------------------------------------------------------------------------
__global__ __launch_bounds__(256)
void transpose_f16_kernel(const float* __restrict__ src,
                          half_t* __restrict__ dstT,
                          float* __restrict__ dstF32,
                          unsigned long long srcBStride,
                          unsigned long long dstTBStride,
                          unsigned long long dstF32BStride) {
    __shared__ float tile[32][33];
    const int b  = blockIdx.z;
    const int c0 = blockIdx.x * 32;
    const int n0 = blockIdx.y * 32;
    const int tx = threadIdx.x & 31;
    const int ty = threadIdx.x >> 5;
    const float* s = src + (size_t)b * srcBStride;
    #pragma unroll
    for (int i = 0; i < 32; i += 8) {
        float v = s[(size_t)(n0 + ty + i) * CC + c0 + tx];
        tile[ty + i][tx] = v;
        if (dstF32)
            dstF32[(size_t)b * dstF32BStride + (size_t)(n0 + ty + i) * CC + c0 + tx] = v;
    }
    __syncthreads();
    half_t* d = dstT + (size_t)b * dstTBStride;
    #pragma unroll
    for (int i = 0; i < 32; i += 8)
        d[(size_t)(c0 + ty + i) * NN + n0 + tx] = (half_t)tile[tx][ty + i];
}

// ---------------------------------------------------------------------------
// One Chebyshev step (software-pipelined f16 WMMA, f32 accumulate):
//   acc[n][c] = sum_m adj[n][m] * zin[m][c]
//   res       = LRC * (degree[n]*zin[n][c] - acc) - zin[n][c]
//   if HAS_PREV: res = 2*res - zprev[n][c]
// Wave computes 16x64; block = 8 waves = 64 rows x 128 cols.  grid = (N/64, B).
// ---------------------------------------------------------------------------
template <bool HAS_PREV>
__global__ __launch_bounds__(256)
void cheb_bmm_kernel(const float* __restrict__ adj,
                     const half_t* __restrict__ zT,      // [B][C][N] f16
                     const float* __restrict__ zin,
                     const float* __restrict__ zprev,
                     const float* __restrict__ degree,
                     float* __restrict__ zout,
                     unsigned long long zStride) {
    const int b    = blockIdx.y;
    const int tid  = threadIdx.x;
    const int wave = tid >> 5;
    const int lane = tid & 31;
    const int lr16 = lane & 15;
    const int hi   = lane >> 4;
    const int row0 = blockIdx.x * 64 + (wave >> 1) * 16;
    const int cH   = (wave & 1) * 64;

    const float*  adjRow = adj + (size_t)b * NN * NN + (size_t)(row0 + lr16) * NN;
    const half_t* zTb    = zT + (size_t)b * CC * NN;
    const int aOff = hi * 8;                          // A layout: lanes>=16 hold K+8
    const int bSel = hi * 8;                          // B layout: lanes>=16 hold K+8

    // A tile loader: adj f32 -> f16, ISA 16-bit A 16x32 layout
    auto loadA = [&](int m) -> v16h {
        v4f a0 = *(const v4f*)(adjRow + m + aOff);
        v4f a1 = *(const v4f*)(adjRow + m + aOff + 4);
        v4f a2 = *(const v4f*)(adjRow + m + 16 + aOff);
        v4f a3 = *(const v4f*)(adjRow + m + 16 + aOff + 4);
        v16h A;
        #pragma unroll
        for (int i = 0; i < 4; ++i) {
            A[i]      = (half_t)a0[i];
            A[4 + i]  = (half_t)a1[i];
            A[8 + i]  = (half_t)a2[i];
            A[12 + i] = (half_t)a3[i];
        }
        return A;
    };
    // B tile loader: per lane 8 f16 at m+bSel and 8 at m+bSel+16 from transposed z
    auto loadB = [&](int m, int t) -> v16h {
        const half_t* bp = zTb + (size_t)(cH + 16 * t + lr16) * NN + m + bSel;
        v8h blo = *(const v8h*)bp;
        v8h bhi = *(const v8h*)(bp + 16);
        return __builtin_shufflevector(blo, bhi,
                 0, 1, 2, 3, 4, 5, 6, 7, 8, 9, 10, 11, 12, 13, 14, 15);
    };

    v8f acc[4] = {v8f{}, v8f{}, v8f{}, v8f{}};

    // ---- 2-stage software pipeline: load (m+32) while WMMAing (m) ----
    v16h A  = loadA(0);
    v16h B0 = loadB(0, 0), B1 = loadB(0, 1), B2 = loadB(0, 2), B3 = loadB(0, 3);

    for (int m = 0; m < NN - 32; m += 32) {
        __builtin_prefetch(adjRow + m + 512, 0, 0);   // stream adj ahead (HBM)
        const int mn = m + 32;
        v16h An  = loadA(mn);
        v16h Bn0 = loadB(mn, 0);
        v16h Bn1 = loadB(mn, 1);
        v16h Bn2 = loadB(mn, 2);
        v16h Bn3 = loadB(mn, 3);
        acc[0] = __builtin_amdgcn_wmma_f32_16x16x32_f16(false, A, false, B0, (short)0, acc[0], false, false);
        acc[1] = __builtin_amdgcn_wmma_f32_16x16x32_f16(false, A, false, B1, (short)0, acc[1], false, false);
        acc[2] = __builtin_amdgcn_wmma_f32_16x16x32_f16(false, A, false, B2, (short)0, acc[2], false, false);
        acc[3] = __builtin_amdgcn_wmma_f32_16x16x32_f16(false, A, false, B3, (short)0, acc[3], false, false);
        A = An; B0 = Bn0; B1 = Bn1; B2 = Bn2; B3 = Bn3;
    }
    acc[0] = __builtin_amdgcn_wmma_f32_16x16x32_f16(false, A, false, B0, (short)0, acc[0], false, false);
    acc[1] = __builtin_amdgcn_wmma_f32_16x16x32_f16(false, A, false, B1, (short)0, acc[1], false, false);
    acc[2] = __builtin_amdgcn_wmma_f32_16x16x32_f16(false, A, false, B2, (short)0, acc[2], false, false);
    acc[3] = __builtin_amdgcn_wmma_f32_16x16x32_f16(false, A, false, B3, (short)0, acc[3], false, false);

    // ---- fused elementwise epilogue on the C/D layout ----
    const float* zinB  = zin  + (size_t)b * zStride;
    float*       zoutB = zout + (size_t)b * zStride;
    const float* degB  = degree + (size_t)b * NN;
    const float* zprB  = HAS_PREV ? (zprev + (size_t)b * zStride) : nullptr;

    #pragma unroll
    for (int v = 0; v < 8; ++v) {
        const int row = row0 + v + 8 * hi;
        const float d = degB[row];
        #pragma unroll
        for (int t = 0; t < 4; ++t) {
            const int col = cH + 16 * t + lr16;
            const size_t off = (size_t)row * CC + col;
            const float z = zinB[off];
            float res = LRC * (d * z - acc[t][v]) - z;
            if (HAS_PREV) res = 2.0f * res - zprB[off];
            zoutB[off] = res;
        }
    }
}

// ---------------------------------------------------------------------------
// Final linear: out[b][i][o] = sum_j Zflat[b][i*384 + j] * W[o][j]
// Full-precision f32 WMMA (16x16x4), two independent accumulate chains.
// Block = 8 waves = 8 col-tiles (128 out cols).  grid = (N/16, B).
// ---------------------------------------------------------------------------
__global__ __launch_bounds__(256)
void final_gemm_kernel(const float* __restrict__ Z,     // [B][K][N][C] contiguous
                       const float* __restrict__ W,     // [C_OUT][384]
                       float* __restrict__ out) {       // [B][N][C_OUT]
    const int b    = blockIdx.y;
    const int tid  = threadIdx.x;
    const int wave = tid >> 5;
    const int lane = tid & 31;
    const int lr16 = lane & 15;
    const int hi   = lane >> 4;
    const int r0   = blockIdx.x * 16;
    const int c0   = wave * 16;

    const float* Arow = Z + (size_t)b * KCH * NN * CC + (size_t)(r0 + lr16) * FDIM;
    const float* Brow = W + (size_t)(c0 + lr16) * FDIM;
    const int koff = hi * 2;                          // f32 A/B: lanes>=16 hold K 2,3

    v8f acc0 = {}, acc1 = {};
    for (int kk = 0; kk < FDIM; kk += 8) {
        v2f a0 = *(const v2f*)(Arow + kk + koff);
        v2f b0 = *(const v2f*)(Brow + kk + koff);
        v2f a1 = *(const v2f*)(Arow + kk + 4 + koff);
        v2f b1 = *(const v2f*)(Brow + kk + 4 + koff);
        acc0 = __builtin_amdgcn_wmma_f32_16x16x4_f32(false, a0, false, b0, (short)0, acc0, false, false);
        acc1 = __builtin_amdgcn_wmma_f32_16x16x4_f32(false, a1, false, b1, (short)0, acc1, false, false);
    }
    v8f acc = acc0 + acc1;                            // D layout is elementwise

    float* op = out + (size_t)b * NN * CC;
    #pragma unroll
    for (int v = 0; v < 8; ++v)
        op[(size_t)(r0 + v + 8 * hi) * CC + c0 + lr16] = acc[v];
}

// ---------------------------------------------------------------------------
extern "C" void kernel_launch(void* const* d_in, const int* in_sizes, int n_in,
                              void* d_out, int out_size, void* d_ws, size_t ws_size,
                              hipStream_t stream) {
    const float* x   = (const float*)d_in[0];   // [B][N][C]
    const float* adj = (const float*)d_in[1];   // [B][N][N]
    const float* W   = (const float*)d_in[2];   // [C_OUT][K*C]
    float* out = (float*)d_out;

    char* ws = (char*)d_ws;
    size_t off = 0;
    auto carve = [&](size_t bytes) {
        void* p = ws + off;
        off = (off + bytes + 255) & ~(size_t)255;
        return p;
    };
    float*  degree  = (float*)carve((size_t)BB * NN * sizeof(float));
    float*  degPart = (float*)carve((size_t)BB * NCHNK * NN * sizeof(float));
    float*  Zbuf    = (float*)carve((size_t)BB * KCH * NN * CC * sizeof(float));
    half_t* xT      = (half_t*)carve((size_t)BB * CC * NN * sizeof(half_t));
    half_t* z1T     = (half_t*)carve((size_t)BB * CC * NN * sizeof(half_t));

    const unsigned long long zStride = (unsigned long long)KCH * NN * CC;

    // 1) degree = column sums of adj (two-stage, 1024-block stream)
    degree_partial_kernel<<<dim3(NN / 256, NCHNK, BB), 256, 0, stream>>>(adj, degPart);
    degree_reduce_kernel<<<dim3(NN / 256, BB), 256, 0, stream>>>(degPart, degree);

    // 2) xT (f16 transposed) + Z0 = x copied into Z buffer
    transpose_f16_kernel<<<dim3(CC / 32, NN / 32, BB), 256, 0, stream>>>(
        x, xT, Zbuf, (unsigned long long)NN * CC, (unsigned long long)CC * NN, zStride);

    // 3) Z1 = LRC*(deg*x - adj@x) - x
    cheb_bmm_kernel<false><<<dim3(NN / 64, BB), 256, 0, stream>>>(
        adj, xT, Zbuf, nullptr, degree, Zbuf + (size_t)NN * CC, zStride);

    // 4) z1T = f16 transpose of Z1
    transpose_f16_kernel<<<dim3(CC / 32, NN / 32, BB), 256, 0, stream>>>(
        Zbuf + (size_t)NN * CC, z1T, nullptr, zStride, (unsigned long long)CC * NN, 0ULL);

    // 5) Z2 = 2*(LRC*(deg*Z1 - adj@Z1) - Z1) - Z0
    cheb_bmm_kernel<true><<<dim3(NN / 64, BB), 256, 0, stream>>>(
        adj, z1T, Zbuf + (size_t)NN * CC, Zbuf, degree,
        Zbuf + (size_t)2 * NN * CC, zStride);

    // 6) out = Zflat @ W^T   (f32 WMMA, exact precision)
    final_gemm_kernel<<<dim3(NN / 16, BB), 256, 0, stream>>>(Zbuf, W, out);
}
